// AttentionHelper_17265768529960
// MI455X (gfx1250) — matrix operand order repaired and checked by reading, not compiled, not run on hardware
//
#include <hip/hip_runtime.h>
#include <hip/hip_bf16.h>

typedef _Float16 v8h  __attribute__((ext_vector_type(8)));
typedef _Float16 v16h __attribute__((ext_vector_type(16)));
typedef float    v8f  __attribute__((ext_vector_type(8)));

constexpr int BB  = 4;     // batch
constexpr int CC  = 64;    // channels (contraction dim of energy)
constexpr int LL  = 4096;  // sequence length
constexpr int QT  = 64;    // q-tile per workgroup (16 per wave x 4 waves)
constexpr int KT  = 64;    // k-tile staged in LDS per iteration
constexpr int PAD = 8;     // halves of padding (keeps 16B alignment, 2-way conflicts max)
constexpr int NT  = 128;   // threads per block (4 wave32)

// join two 16B LDS vectors into one WMMA operand (8 VGPRs of f16)
__device__ __forceinline__ v16h join8(const _Float16* lo, const _Float16* hi) {
    v8h a = *(const v8h*)lo;
    v8h b = *(const v8h*)hi;
    return __builtin_shufflevector(a, b, 0,1,2,3,4,5,6,7,8,9,10,11,12,13,14,15);
}

__global__ __launch_bounds__(NT)
void attn_fused(const float* __restrict__ Qp, const float* __restrict__ Kp,
                const float* __restrict__ Vp, const float* __restrict__ Mp,
                float* __restrict__ outp, float* __restrict__ attp)
{
    __shared__ _Float16 Qt[QT][CC + PAD];   // Q transposed: [q][c]
    __shared__ _Float16 Kt[KT][CC + PAD];   // K transposed: [k][c]
    __shared__ _Float16 Vs[CC][KT + PAD];   // V row-major : [c][k]
    __shared__ float    lmS[KT];            // log(mask + 1e-6)
    __shared__ float    mkS[KT];            // mask

    const int tid  = threadIdx.x;
    const int wave = tid >> 5;
    const int lane = tid & 31;
    const int n    = lane & 15;   // column index inside 16x16 tile
    const int h    = lane >> 4;   // lane-half selector

    const int b    = blockIdx.x / (LL / QT);
    const int qblk = (blockIdx.x % (LL / QT)) * QT;
    const int qrow = 16 * wave + n;          // this lane's q column (local)
    const int qcol = qblk + qrow;            // global q for S/attention stores

    const float scale = 0.125f;              // 1/sqrt(64)
    const size_t bCL = (size_t)b * CC * LL;
    const float* Qb = Qp + bCL;
    const float* Kb = Kp + bCL;
    const float* Vb = Vp + bCL;
    const float* Mb = Mp + (size_t)b * LL;
    float* att = attp + (size_t)b * LL * LL;

    const v8f zero = {0.f,0.f,0.f,0.f,0.f,0.f,0.f,0.f};

    // ---- stage Q tile transposed into LDS (coalesced reads along L) ----
    for (int idx = tid; idx < CC * QT; idx += NT) {
        int j = idx & (QT - 1);
        int c = idx >> 6;
        Qt[j][c] = (_Float16)Qb[(size_t)c * LL + qblk + j];
    }
    __syncthreads();

    // B operands (Q) are fixed for the whole kernel: kdim=c, n=q
    v16h bq0 = join8(&Qt[qrow][ 0 + 8*h], &Qt[qrow][16 + 8*h]);
    v16h bq1 = join8(&Qt[qrow][32 + 8*h], &Qt[qrow][48 + 8*h]);

    // ================= Pass 1: online softmax statistics =================
    float m = -3.0e38f, l = 0.0f;
    for (int kt = 0; kt < LL; kt += KT) {
        __syncthreads();
        for (int idx = tid; idx < CC * KT; idx += NT) {
            int j = idx & (KT - 1);
            int c = idx >> 6;
            Kt[j][c] = (_Float16)Kb[(size_t)c * LL + kt + j];
        }
        if (tid < KT) lmS[tid] = __logf(Mb[kt + tid] + 1e-6f);
        __syncthreads();

        for (int ks = 0; ks < KT; ks += 16) {
            // A operand = K^T rows: M=k, kdim=c (two 32-c chunks)
            v16h ak0 = join8(&Kt[ks + n][ 0 + 8*h], &Kt[ks + n][16 + 8*h]);
            v16h ak1 = join8(&Kt[ks + n][32 + 8*h], &Kt[ks + n][48 + 8*h]);
            v8f s = __builtin_amdgcn_wmma_f32_16x16x32_f16(false, ak0, false, bq0, (short)0, zero, false, false);
            s     = __builtin_amdgcn_wmma_f32_16x16x32_f16(false, ak1, false, bq1, (short)0, s,    false, false);

            float sv[8];
            float tmax = -3.0e38f;
            #pragma unroll
            for (int r = 0; r < 8; ++r) {
                sv[r] = s[r] * scale + lmS[ks + r + 8*h];
                tmax = fmaxf(tmax, sv[r]);
            }
            tmax = fmaxf(tmax, __shfl_xor(tmax, 16, 32)); // lanes L/L+16 share q
            float nm = fmaxf(m, tmax);
            float ps = 0.0f;
            #pragma unroll
            for (int r = 0; r < 8; ++r) ps += __expf(sv[r] - nm);
            ps += __shfl_xor(ps, 16, 32);
            l = l * __expf(m - nm) + ps;
            m = nm;
        }
    }
    const float invl = 1.0f / l;

    // ====== Pass 2: recompute S, write normalized P^T, accumulate out ======
    v8f acc0 = zero, acc1 = zero, acc2 = zero, acc3 = zero;

    for (int kt = 0; kt < LL; kt += KT) {
        __syncthreads();
        for (int idx = tid; idx < CC * KT; idx += NT) {
            int j = idx & (KT - 1);
            int c = idx >> 6;
            float kv = Kb[(size_t)c * LL + kt + j];
            float vv = Vb[(size_t)c * LL + kt + j];
            Kt[j][c] = (_Float16)kv;
            Vs[c][j] = (_Float16)vv;
        }
        if (tid < KT) {
            float mv = Mb[kt + tid];
            lmS[tid] = __logf(mv + 1e-6f);
            mkS[tid] = mv;
        }
        __syncthreads();

        for (int ks = 0; ks < KT; ks += 32) {
            // two 16x16 S^T subtiles (k = ks..ks+15 and ks+16..ks+31)
            v16h aA0 = join8(&Kt[ks +      n][ 0 + 8*h], &Kt[ks +      n][16 + 8*h]);
            v16h aA1 = join8(&Kt[ks +      n][32 + 8*h], &Kt[ks +      n][48 + 8*h]);
            v8f sA = __builtin_amdgcn_wmma_f32_16x16x32_f16(false, aA0, false, bq0, (short)0, zero, false, false);
            sA     = __builtin_amdgcn_wmma_f32_16x16x32_f16(false, aA1, false, bq1, (short)0, sA,   false, false);
            v16h aB0 = join8(&Kt[ks + 16 + n][ 0 + 8*h], &Kt[ks + 16 + n][16 + 8*h]);
            v16h aB1 = join8(&Kt[ks + 16 + n][32 + 8*h], &Kt[ks + 16 + n][48 + 8*h]);
            v8f sB = __builtin_amdgcn_wmma_f32_16x16x32_f16(false, aB0, false, bq0, (short)0, zero, false, false);
            sB     = __builtin_amdgcn_wmma_f32_16x16x32_f16(false, aB1, false, bq1, (short)0, sB,   false, false);

            // normalize, store P^T coalesced (non-temporal: write-once 268MB
            // stream must not evict the K/V working set from L2), and pack the
            // A operand for the out-WMMA. The S^T tile layout lines up with the
            // A(16x32) per-lane layout: pure per-lane packing, no shuffles.
            v16h ap;
            #pragma unroll
            for (int r = 0; r < 8; ++r) {
                int kA = ks + r + 8*h;
                float pA = __expf(sA[r] * scale + lmS[kA] - m) * invl * mkS[kA];
                __builtin_nontemporal_store(pA, &att[(size_t)(kt + kA) * LL + qcol]);
                ap[r] = (_Float16)pA;
                int kB = kA + 16;
                float pB = __expf(sB[r] * scale + lmS[kB] - m) * invl * mkS[kB];
                __builtin_nontemporal_store(pB, &att[(size_t)(kt + kB) * LL + qcol]);
                ap[8 + r] = (_Float16)pB;
            }

            // B operand = V^T: kdim=k (32), n = c (4 chunks of 16 channels)
            v16h bv0 = join8(&Vs[ 0 + n][ks + 8*h], &Vs[ 0 + n][ks + 16 + 8*h]);
            v16h bv1 = join8(&Vs[16 + n][ks + 8*h], &Vs[16 + n][ks + 16 + 8*h]);
            v16h bv2 = join8(&Vs[32 + n][ks + 8*h], &Vs[32 + n][ks + 16 + 8*h]);
            v16h bv3 = join8(&Vs[48 + n][ks + 8*h], &Vs[48 + n][ks + 16 + 8*h]);
            acc0 = __builtin_amdgcn_wmma_f32_16x16x32_f16(false, ap, false, bv0, (short)0, acc0, false, false);
            acc1 = __builtin_amdgcn_wmma_f32_16x16x32_f16(false, ap, false, bv1, (short)0, acc1, false, false);
            acc2 = __builtin_amdgcn_wmma_f32_16x16x32_f16(false, ap, false, bv2, (short)0, acc2, false, false);
            acc3 = __builtin_amdgcn_wmma_f32_16x16x32_f16(false, ap, false, bv3, (short)0, acc3, false, false);
        }
    }

    // out[b][c][q]: D_out tile has M=q (reg r + 8h), N=c (lane n), 4 c-chunks
    {
        const int qb = qblk + 16 * wave + 8 * h;
        #pragma unroll
        for (int r = 0; r < 8; ++r) {
            __builtin_nontemporal_store(acc0[r], &outp[bCL + (size_t)( 0 + n) * LL + qb + r]);
            __builtin_nontemporal_store(acc1[r], &outp[bCL + (size_t)(16 + n) * LL + qb + r]);
            __builtin_nontemporal_store(acc2[r], &outp[bCL + (size_t)(32 + n) * LL + qb + r]);
            __builtin_nontemporal_store(acc3[r], &outp[bCL + (size_t)(48 + n) * LL + qb + r]);
        }
    }
}

extern "C" void kernel_launch(void* const* d_in, const int* in_sizes, int n_in,
                              void* d_out, int out_size, void* d_ws, size_t ws_size,
                              hipStream_t stream) {
    const float* Q = (const float*)d_in[0];
    const float* K = (const float*)d_in[1];
    const float* V = (const float*)d_in[2];
    const float* M = (const float*)d_in[3];
    float* out = (float*)d_out;                       // [B, C, L]
    float* att = out + (size_t)BB * CC * LL;          // [B, L, L] (attention_t)

    dim3 grid(BB * (LL / QT));   // 256 workgroups
    dim3 block(NT);              // 128 threads = 4 wave32
    hipLaunchKernelGGL(attn_fused, grid, block, 0, stream, Q, K, V, M, out, att);
}
